// GeniePathLazy_901943132751
// MI455X (gfx1250) — matrix operand order repaired and checked by reading, not compiled, not run on hardware
//
#include <hip/hip_runtime.h>
#include <hip/hip_bf16.h>
#include <math.h>

// ---------------------------------------------------------------------------
// GeniePath-Lazy on gfx1250: bf16 WMMA GEMMs (pre-converted bf16 activations,
// dual-tile accumulation per wave, scalar-uniform tile control so WMMA never
// executes under a modified EXEC) + scalar GAT scatter-softmax.
// ---------------------------------------------------------------------------

typedef __attribute__((ext_vector_type(16))) __bf16 v16bf;
typedef __attribute__((ext_vector_type(8)))  float  v8f;

#define IN_DIM 256
#define DIM    256
#define HID    256
#define OUT_DIM 64
#define LAYERS 4
#define G4H    1024   // 4*HID
#define K2D    512    // 2*DIM

// ---- bf16 helpers (storage = unsigned short) ------------------------------
__device__ __forceinline__ unsigned short f2bfu(float f) {
    unsigned u = __float_as_uint(f);
    u = u + 0x7FFFu + ((u >> 16) & 1u);          // round-to-nearest-even
    return (unsigned short)(u >> 16);
}
__device__ __forceinline__ __bf16 bfraw(unsigned short s) {
    __bf16 t; __builtin_memcpy(&t, &s, 2); return t;
}

// A fragment (16x32 bf16 layout): lane holds row lane&15;
// elems 0-7 = K[hi*8 .. +7], elems 8-15 = K[16+hi*8 .. +7]
__device__ __forceinline__ v16bf mk_a(const unsigned short* row, int k, int hi) {
    v16bf a;
    const unsigned short* p = row + k + hi * 8;
#pragma unroll
    for (int j = 0; j < 8; ++j) {
        a[j]     = bfraw(p[j]);
        a[8 + j] = bfraw(p[16 + j]);
    }
    return a;
}
// B fragment (32x16 bf16 layout): lane holds col lane&15;
// elems 0-7 = K[hi*8 .. +7], elems 8-15 = K[16+hi*8 .. +7]
__device__ __forceinline__ v16bf mk_b(const unsigned short* col, int k, int hi) {
    v16bf b;
    const unsigned short* p = col + k + hi * 8;
#pragma unroll
    for (int j = 0; j < 8; ++j) {
        b[j]     = bfraw(p[j]);
        b[8 + j] = bfraw(p[16 + j]);
    }
    return b;
}

// ---- weight / activation prep ---------------------------------------------
// Bp[n*K + k] = bf16( W[k*N + n] )   (transpose + convert)
__global__ void conv_transpose_bf16(const float* __restrict__ W,
                                    unsigned short* __restrict__ Bp,
                                    int K, int N) {
    int idx = blockIdx.x * blockDim.x + threadIdx.x;
    if (idx >= K * N) return;
    int n = idx / K, k = idx - n * K;
    Bp[idx] = f2bfu(W[(size_t)k * N + n]);
}
// straight convert (row-contiguous already)
__global__ void conv_bf16(const float* __restrict__ W,
                          unsigned short* __restrict__ Bp, int total) {
    int idx = blockIdx.x * blockDim.x + threadIdx.x;
    if (idx < total) Bp[idx] = f2bfu(W[idx]);
}

__global__ void fill_f32(float* p, float v, int n) {
    int i = blockIdx.x * blockDim.x + threadIdx.x;
    if (i < n) p[i] = v;
}
__global__ void fill_u32(unsigned* p, unsigned v, int n) {
    int i = blockIdx.x * blockDim.x + threadIdx.x;
    if (i < n) p[i] = v;
}

// ---- GEMM: C/Cb[M,N] = act( A(bf16)[M,K] * Bp(bf16 packed) + bias ) -------
// block = 128 threads = 4 waves; each wave: two 16x16 tiles sharing A frag.
// Tile control flow is forced scalar (readfirstlane) so WMMA always runs
// with EXEC all-1s as the ISA requires.
__global__ void __launch_bounds__(128)
gemm_bf16(const unsigned short* __restrict__ A,
          const unsigned short* __restrict__ Bp,
          const float* __restrict__ bias,
          float* __restrict__ C, unsigned short* __restrict__ Cb,
          int M, int N, int K, int act) {
    int lane = threadIdx.x & 31;
    int n0 = __builtin_amdgcn_readfirstlane(blockIdx.x * 128 + (threadIdx.x >> 5) * 32);
    if (n0 >= N) return;                       // scalar branch
    bool two = (n0 + 16) < N;                  // scalar condition
    int m0 = blockIdx.y * 16;
    int ml = lane & 15, hi = lane >> 4;
    int m = m0 + ml; if (m >= M) m = M - 1;
    const unsigned short* arow  = A  + (size_t)m * K;
    const unsigned short* bcol0 = Bp + (size_t)(n0 + ml) * K;
    const unsigned short* bcol1 = Bp + (size_t)(n0 + 16 + ml) * K;
    v8f acc0 = {}, acc1 = {};
    for (int k = 0; k < K; k += 32) {
        __builtin_prefetch(arow + k + 32, 0, 3);   // speculative, safe past end
        v16bf a  = mk_a(arow, k, hi);
        v16bf b0 = mk_b(bcol0, k, hi);
        acc0 = __builtin_amdgcn_wmma_f32_16x16x32_bf16(false, a, false, b0,
                                                       (short)0, acc0, false, false);
        if (two) {
            v16bf b1 = mk_b(bcol1, k, hi);
            acc1 = __builtin_amdgcn_wmma_f32_16x16x32_bf16(false, a, false, b1,
                                                           (short)0, acc1, false, false);
        }
    }
#pragma unroll
    for (int t = 0; t < 2; ++t) {
        int nc = n0 + t * 16 + ml;
        if (n0 + t * 16 >= N) break;           // scalar
        v8f acc = t ? acc1 : acc0;
        float bv = bias ? bias[nc] : 0.f;
#pragma unroll
        for (int r = 0; r < 8; ++r) {
            int row = m0 + hi * 8 + r;
            if (row < M) {
                float v = acc[r] + bv;
                if (act == 1) v = tanhf(v);
                if (C)  C [(size_t)row * N + nc] = v;
                if (Cb) Cb[(size_t)row * N + nc] = f2bfu(v);
            }
        }
    }
}

// ---- fused LSTM gates:
// G[M,1024] = htmp·wih[:, :256]ᵀ + xc·wih[:,256:]ᵀ + h·whhᵀ  (all A bf16)
__global__ void __launch_bounds__(128)
lstm_gates(const unsigned short* __restrict__ Htmp,
           const unsigned short* __restrict__ Xc,
           const unsigned short* __restrict__ Hprev,
           const unsigned short* __restrict__ Wih,   // [1024][512] bf16
           const unsigned short* __restrict__ Whh,   // [1024][256] bf16
           float* __restrict__ Gates, int M) {
    int lane = threadIdx.x & 31;
    int n0 = __builtin_amdgcn_readfirstlane(blockIdx.x * 128 + (threadIdx.x >> 5) * 32);
    int m0 = blockIdx.y * 16;
    int ml = lane & 15, hi = lane >> 4;
    int m = m0 + ml; if (m >= M) m = M - 1;
    const unsigned short* h1 = Htmp  + (size_t)m * DIM;
    const unsigned short* x1 = Xc    + (size_t)m * DIM;
    const unsigned short* hp = Hprev + (size_t)m * HID;
    const unsigned short* wih0 = Wih + (size_t)(n0 + ml) * K2D;
    const unsigned short* wih1 = Wih + (size_t)(n0 + 16 + ml) * K2D;
    const unsigned short* whh0 = Whh + (size_t)(n0 + ml) * HID;
    const unsigned short* whh1 = Whh + (size_t)(n0 + 16 + ml) * HID;
    v8f acc0 = {}, acc1 = {};
    for (int k = 0; k < DIM; k += 32) {
        v16bf a0 = mk_a(h1, k, hi);
        acc0 = __builtin_amdgcn_wmma_f32_16x16x32_bf16(false, a0, false, mk_b(wih0, k, hi),
                                                       (short)0, acc0, false, false);
        acc1 = __builtin_amdgcn_wmma_f32_16x16x32_bf16(false, a0, false, mk_b(wih1, k, hi),
                                                       (short)0, acc1, false, false);
        v16bf a1 = mk_a(x1, k, hi);
        acc0 = __builtin_amdgcn_wmma_f32_16x16x32_bf16(false, a1, false, mk_b(wih0, 256 + k, hi),
                                                       (short)0, acc0, false, false);
        acc1 = __builtin_amdgcn_wmma_f32_16x16x32_bf16(false, a1, false, mk_b(wih1, 256 + k, hi),
                                                       (short)0, acc1, false, false);
        v16bf a2 = mk_a(hp, k, hi);
        acc0 = __builtin_amdgcn_wmma_f32_16x16x32_bf16(false, a2, false, mk_b(whh0, k, hi),
                                                       (short)0, acc0, false, false);
        acc1 = __builtin_amdgcn_wmma_f32_16x16x32_bf16(false, a2, false, mk_b(whh1, k, hi),
                                                       (short)0, acc1, false, false);
    }
#pragma unroll
    for (int t = 0; t < 2; ++t) {
        v8f acc = t ? acc1 : acc0;
        int nc = n0 + t * 16 + ml;
#pragma unroll
        for (int r = 0; r < 8; ++r) {
            int row = m0 + hi * 8 + r;
            if (row < M) Gates[(size_t)row * G4H + nc] = acc[r];
        }
    }
}

// ---- GAT attention scalars: a_s = xw·a_src, a_d = xw·a_dst (wave/node) ----
__global__ void attn_dots(const float* __restrict__ XW,
                          const float* __restrict__ asrc,
                          const float* __restrict__ adst,
                          float* __restrict__ as_o, float* __restrict__ ad_o, int M) {
    int wave = threadIdx.x >> 5, lane = threadIdx.x & 31;
    int node = blockIdx.x * (blockDim.x >> 5) + wave;
    if (node >= M) return;
    const float* row = XW + (size_t)node * DIM;
    float s0 = 0.f, s1 = 0.f;
    for (int k = lane; k < DIM; k += 32) {
        float v = row[k];
        s0 += v * asrc[k];
        s1 += v * adst[k];
    }
#pragma unroll
    for (int off = 16; off > 0; off >>= 1) {
        s0 += __shfl_down(s0, off, 32);
        s1 += __shfl_down(s1, off, 32);
    }
    if (lane == 0) { as_o[node] = s0; ad_o[node] = s1; }
}

// monotonic float<->uint mapping for atomicMax over floats
__device__ __forceinline__ unsigned ford(float f) {
    unsigned u = __float_as_uint(f);
    return (u & 0x80000000u) ? ~u : (u | 0x80000000u);
}
__device__ __forceinline__ float funord(unsigned u) {
    return (u & 0x80000000u) ? __uint_as_float(u & 0x7FFFFFFFu)
                             : __uint_as_float(~u);
}

// pass 1: alpha = leaky(a_s[src]+a_d[dst]); segment-max via atomicMax(bits)
__global__ void edge_alpha_max(const int* __restrict__ srcA, const int* __restrict__ dstA,
                               const float* __restrict__ as_, const float* __restrict__ ad_,
                               float* __restrict__ alpha, unsigned* __restrict__ mbits,
                               int E, int Nn) {
    int e = blockIdx.x * blockDim.x + threadIdx.x;
    int ET = E + Nn;
    if (e >= ET) return;
    int s = (e < E) ? srcA[e] : (e - E);
    int d = (e < E) ? dstA[e] : (e - E);
    float a = as_[s] + ad_[d];
    a = (a > 0.f) ? a : 0.2f * a;
    alpha[e] = a;
    atomicMax(&mbits[d], ford(a));
}
// pass 2: e = exp(alpha - m[dst]); denom[dst] += e
__global__ void edge_exp_sum(const int* __restrict__ srcA, const int* __restrict__ dstA,
                             const float* __restrict__ alpha, const unsigned* __restrict__ mbits,
                             float* __restrict__ eexp, float* __restrict__ denom,
                             int E, int Nn) {
    int e = blockIdx.x * blockDim.x + threadIdx.x;
    int ET = E + Nn;
    if (e >= ET) return;
    int d = (e < E) ? dstA[e] : (e - E);
    float ev = expf(alpha[e] - funord(mbits[d]));
    eexp[e] = ev;
    atomicAdd(&denom[d], ev);
}
// pass 3: Out[dst] += (e/denom[dst]) * XW[src]   (64 threads/edge, 4 ch each)
__global__ void edge_scatter(const int* __restrict__ srcA, const int* __restrict__ dstA,
                             const float* __restrict__ eexp, const float* __restrict__ denom,
                             const float* __restrict__ XW, float* __restrict__ Out,
                             int E, int Nn) {
    int t = blockIdx.x * blockDim.x + threadIdx.x;
    int eid = t >> 6;
    int ch = (t & 63) * 4;
    int ET = E + Nn;
    if (eid >= ET) return;
    int s = (eid < E) ? srcA[eid] : (eid - E);
    int d = (eid < E) ? dstA[eid] : (eid - E);
    float coef = eexp[eid] / denom[d];
    const float* xr = XW + (size_t)s * DIM + ch;
    float* orow = Out + (size_t)d * DIM + ch;
#pragma unroll
    for (int j = 0; j < 4; ++j) atomicAdd(&orow[j], coef * xr[j]);
}

// h_tmp(bf16) = tanh(out + b)
__global__ void bias_tanh_bf16(const float* __restrict__ X, const float* __restrict__ b,
                               unsigned short* __restrict__ Xb, int total) {
    int i = blockIdx.x * blockDim.x + threadIdx.x;
    if (i < total) Xb[i] = f2bfu(tanhf(X[i] + b[i & (DIM - 1)]));
}

// LSTM pointwise cell (torch gate order i,f,g,o); h kept only as bf16 mirror
__global__ void lstm_cell(const float* __restrict__ G, float* __restrict__ c,
                          unsigned short* __restrict__ hb, int total) {
    int i = blockIdx.x * blockDim.x + threadIdx.x;
    if (i >= total) return;
    int n = i >> 8, k = i & 255;
    const float* g = G + (size_t)n * G4H;
    float ig = g[k], fg = g[HID + k], gg = g[2 * HID + k], og = g[3 * HID + k];
    float si = 1.f / (1.f + expf(-ig));
    float sf = 1.f / (1.f + expf(-fg));
    float so = 1.f / (1.f + expf(-og));
    float cn = sf * c[i] + si * tanhf(gg);
    c[i] = cn;
    hb[i] = f2bfu(so * tanhf(cn));
}

// ---------------------------------------------------------------------------
extern "C" void kernel_launch(void* const* d_in, const int* in_sizes, int n_in,
                              void* d_out, int out_size, void* d_ws, size_t ws_size,
                              hipStream_t stream) {
    const float* x        = (const float*)d_in[0];
    const int*   eidx     = (const int*)d_in[1];
    const float* lin1_w   = (const float*)d_in[2];
    const float* lin1_b   = (const float*)d_in[3];
    const float* gat_w    = (const float*)d_in[4];
    const float* gat_asrc = (const float*)d_in[5];
    const float* gat_adst = (const float*)d_in[6];
    const float* gat_b    = (const float*)d_in[7];
    const float* lstm_wih = (const float*)d_in[8];
    const float* lstm_whh = (const float*)d_in[9];
    const float* lin2_w   = (const float*)d_in[10];
    const float* lin2_b   = (const float*)d_in[11];

    const int NN = in_sizes[0] / IN_DIM;   // 10000
    const int E  = in_sizes[1] / 2;        // 320000
    const int ET = E + NN;                 // +self loops
    const int* srcA = eidx;
    const int* dstA = eidx + E;

    // ---- workspace carve-up ----
    char* p = (char*)d_ws;
    auto alloc = [&](size_t bytes) -> void* {
        void* r = (void*)p;
        p += (bytes + 255) & ~(size_t)255;
        return r;
    };
    // bf16 weights (packed B: row n = column n of effective B, contiguous K)
    unsigned short* lin1T = (unsigned short*)alloc((size_t)IN_DIM * DIM * 2);
    unsigned short* gatT  = (unsigned short*)alloc((size_t)LAYERS * DIM * DIM * 2);
    unsigned short* wihB  = (unsigned short*)alloc((size_t)LAYERS * G4H * K2D * 2);
    unsigned short* whhB  = (unsigned short*)alloc((size_t)LAYERS * G4H * HID * 2);
    unsigned short* lin2T = (unsigned short*)alloc((size_t)DIM * OUT_DIM * 2);
    // bf16 activations (GEMM A operands)
    unsigned short* xb    = (unsigned short*)alloc((size_t)NN * IN_DIM * 2);
    unsigned short* x0b   = (unsigned short*)alloc((size_t)NN * DIM * 2);
    unsigned short* htmpb = (unsigned short*)alloc((size_t)NN * DIM * 2);
    unsigned short* hb    = (unsigned short*)alloc((size_t)NN * HID * 2);
    // f32 buffers
    float* xw    = (float*)alloc((size_t)NN * DIM * 4);
    float* htmp  = (float*)alloc((size_t)NN * DIM * 4);   // GAT scatter target
    float* gates = (float*)alloc((size_t)NN * G4H * 4);
    float* cbuf  = (float*)alloc((size_t)NN * HID * 4);
    float* as_   = (float*)alloc((size_t)NN * 4);
    float* ad_   = (float*)alloc((size_t)NN * 4);
    float* alpha = (float*)alloc((size_t)ET * 4);
    float* eexp  = (float*)alloc((size_t)ET * 4);
    float* denom = (float*)alloc((size_t)NN * 4);
    unsigned* mbits = (unsigned*)alloc((size_t)NN * 4);
    (void)ws_size; (void)n_in; (void)out_size;

    const int TPB = 256;
    auto cdiv = [](int a, int b) { return (a + b - 1) / b; };

    // ---- weight prep ----
    conv_transpose_bf16<<<cdiv(IN_DIM * DIM, TPB), TPB, 0, stream>>>(lin1_w, lin1T, IN_DIM, DIM);
    for (int l = 0; l < LAYERS; ++l)
        conv_transpose_bf16<<<cdiv(DIM * DIM, TPB), TPB, 0, stream>>>(
            gat_w + (size_t)l * DIM * DIM, gatT + (size_t)l * DIM * DIM, DIM, DIM);
    conv_bf16<<<cdiv(LAYERS * G4H * K2D, TPB), TPB, 0, stream>>>(lstm_wih, wihB, LAYERS * G4H * K2D);
    conv_bf16<<<cdiv(LAYERS * G4H * HID, TPB), TPB, 0, stream>>>(lstm_whh, whhB, LAYERS * G4H * HID);
    conv_transpose_bf16<<<cdiv(DIM * OUT_DIM, TPB), TPB, 0, stream>>>(lin2_w, lin2T, DIM, OUT_DIM);

    // ---- activation prep / state init ----
    conv_bf16<<<cdiv(NN * IN_DIM, TPB), TPB, 0, stream>>>(x, xb, NN * IN_DIM);
    fill_u32<<<cdiv(NN * HID / 2, TPB), TPB, 0, stream>>>((unsigned*)hb, 0u, NN * HID / 2);
    fill_f32<<<cdiv(NN * HID, TPB), TPB, 0, stream>>>(cbuf, 0.f, NN * HID);

    // ---- x0 = x @ lin1_w + lin1_b  (bf16 mirror only; x0 f32 never needed) ----
    {
        dim3 g(cdiv(DIM, 128), cdiv(NN, 16));
        gemm_bf16<<<g, 128, 0, stream>>>(xb, lin1T, lin1_b, nullptr, x0b,
                                         NN, DIM, IN_DIM, 0);
    }

    // ---- per-layer: GAT (from x0) then LSTM step ----
    for (int l = 0; l < LAYERS; ++l) {
        dim3 g(cdiv(DIM, 128), cdiv(NN, 16));
        gemm_bf16<<<g, 128, 0, stream>>>(x0b, gatT + (size_t)l * DIM * DIM,
                                         nullptr, xw, nullptr, NN, DIM, DIM, 0);
        attn_dots<<<cdiv(NN, 8), 256, 0, stream>>>(xw, gat_asrc + l * DIM,
                                                   gat_adst + l * DIM, as_, ad_, NN);
        fill_u32<<<cdiv(NN, TPB), TPB, 0, stream>>>(mbits, 0u, NN);
        fill_f32<<<cdiv(NN, TPB), TPB, 0, stream>>>(denom, 0.f, NN);
        fill_f32<<<cdiv(NN * DIM, TPB), TPB, 0, stream>>>(htmp, 0.f, NN * DIM);
        edge_alpha_max<<<cdiv(ET, TPB), TPB, 0, stream>>>(srcA, dstA, as_, ad_, alpha, mbits, E, NN);
        edge_exp_sum<<<cdiv(ET, TPB), TPB, 0, stream>>>(srcA, dstA, alpha, mbits, eexp, denom, E, NN);
        edge_scatter<<<cdiv(ET * 64, TPB), TPB, 0, stream>>>(srcA, dstA, eexp, denom, xw, htmp, E, NN);
        bias_tanh_bf16<<<cdiv(NN * DIM, TPB), TPB, 0, stream>>>(htmp, gat_b + l * DIM,
                                                                htmpb, NN * DIM);

        const unsigned short* xcb = (l == 0) ? x0b : hb;
        dim3 gg(cdiv(G4H, 128), cdiv(NN, 16));
        lstm_gates<<<gg, 128, 0, stream>>>(htmpb, xcb, hb,
                                           wihB + (size_t)l * G4H * K2D,
                                           whhB + (size_t)l * G4H * HID, gates, NN);
        lstm_cell<<<cdiv(NN * HID, TPB), TPB, 0, stream>>>(gates, cbuf, hb, NN * HID);
    }

    // ---- out = h @ lin2_w + lin2_b ----
    {
        dim3 g(cdiv(OUT_DIM, 128), cdiv(NN, 16));
        gemm_bf16<<<g, 128, 0, stream>>>(hb, lin2T, lin2_b, (float*)d_out, nullptr,
                                         NN, OUT_DIM, DIM, 0);
    }
}
